// BaichuanAttention_18932215840895
// MI455X (gfx1250) — compile-verified, matrix operand order checked by reading, not run on hardware
//
#include <hip/hip_runtime.h>
#include <hip/hip_bf16.h>
#include <cstdint>

// ---------------------------------------------------------------------------
// Baichuan attention forward for MI455X (gfx1250, wave32, WMMA).
//   B=1, S=2048, E=4096, H=32, D=128
// Strategy: fp32 weights are streamed once and converted to bf16 in LDS
// (TF32-style), all matmuls use v_wmma_f32_16x16x32_bf16 with f32 accum.
// Attention K/V tiles are moved with global_load_async_to_lds_b128.
// ---------------------------------------------------------------------------

typedef __attribute__((ext_vector_type(16))) __bf16 v16bf;
typedef __attribute__((ext_vector_type(8)))  __bf16 v8bf;
typedef __attribute__((ext_vector_type(8)))  float  v8f;

constexpr int SEQ = 2048;
constexpr int EMB = 4096;
constexpr int NH  = 32;
constexpr int HD  = 128;
constexpr float SCALE = 0.08838834764831845f; // 1/sqrt(128)
constexpr float LOG2E = 1.4426950408889634f;

// ---------------------------------------------------------------------------
// WMMA fragment load: 16x32 bf16 A (or B-as-Nt) fragment from an LDS tile.
// Lane L in [0,15]:  row = L,    K = {0..7, 16..23}
// Lane L in [16,31]: row = L-16, K = {8..15, 24..31}
// rowp points at (row, kk_base); kb = (lane>>4)*8.
// ---------------------------------------------------------------------------
__device__ __forceinline__ v16bf ldfrag(const __bf16* rowp, int kb) {
    union { v16bf v; v8bf h[2]; } u;
    u.h[0] = *(const v8bf*)(rowp + kb);
    u.h[1] = *(const v8bf*)(rowp + kb + 16);
    return u.v;
}

__device__ __forceinline__ v8f wmma_bf16(v16bf a, v16bf b, v8f c) {
    return __builtin_amdgcn_wmma_f32_16x16x32_bf16(
        /*neg_a=*/false, a, /*neg_b=*/false, b,
        /*c_mod=*/(short)0, c, /*reuse_a=*/false, /*reuse_b=*/false);
}

// ---------------------------------------------------------------------------
// GEMM: C[M,N] = A[M,K] * B[N,K]^T   (both row-major, K contiguous)
// Block tile 128x128x32, 8 waves (256 threads), wave tile 32x64.
// LDS tiles stored bf16 with padded stride 40 elements.
// ---------------------------------------------------------------------------
constexpr int TSTR = 40; // padded LDS row stride (elements)

__device__ __forceinline__ void tile_load_f32(__bf16* dst, const float* src,
                                              size_t row0, int k0, int ld, int tid) {
    int r = tid >> 1;
    int c = (tid & 1) * 16;
    const float4* g = (const float4*)(src + (row0 + r) * (size_t)ld + k0 + c);
    float4 f0 = g[0], f1 = g[1], f2 = g[2], f3 = g[3];
    v8bf h0, h1;
    h0[0]=(__bf16)f0.x; h0[1]=(__bf16)f0.y; h0[2]=(__bf16)f0.z; h0[3]=(__bf16)f0.w;
    h0[4]=(__bf16)f1.x; h0[5]=(__bf16)f1.y; h0[6]=(__bf16)f1.z; h0[7]=(__bf16)f1.w;
    h1[0]=(__bf16)f2.x; h1[1]=(__bf16)f2.y; h1[2]=(__bf16)f2.z; h1[3]=(__bf16)f2.w;
    h1[4]=(__bf16)f3.x; h1[5]=(__bf16)f3.y; h1[6]=(__bf16)f3.z; h1[7]=(__bf16)f3.w;
    *(v8bf*)(dst + r * TSTR + c)     = h0;
    *(v8bf*)(dst + r * TSTR + c + 8) = h1;
}

__device__ __forceinline__ void tile_load_bf16(__bf16* dst, const __bf16* src,
                                               size_t row0, int k0, int ld, int tid) {
    int r = tid >> 1;
    int c = (tid & 1) * 16;
    const v8bf* g = (const v8bf*)(src + (row0 + r) * (size_t)ld + k0 + c);
    *(v8bf*)(dst + r * TSTR + c)     = g[0];
    *(v8bf*)(dst + r * TSTR + c + 8) = g[1];
}

template <bool AF32, bool BF32, bool CF32>
__global__ void __launch_bounds__(256)
gemm_bt_kernel(const void* __restrict__ Ap, const void* __restrict__ Bp,
               void* __restrict__ Cp, int M, int N, int K) {
    __shared__ __bf16 As[128 * TSTR];
    __shared__ __bf16 Bs[128 * TSTR];

    const int tid  = threadIdx.x;
    const int lane = tid & 31;
    const int wid  = tid >> 5;
    const int wm   = (wid & 3) * 32;   // 4 waves along M
    const int wn   = (wid >> 2) * 64;  // 2 waves along N
    const int r    = lane & 15;
    const int kb   = (lane >> 4) * 8;
    const size_t m0 = (size_t)blockIdx.y * 128;
    const size_t n0 = (size_t)blockIdx.x * 128;

    v8f acc[2][4];
#pragma unroll
    for (int mi = 0; mi < 2; ++mi)
#pragma unroll
        for (int j = 0; j < 4; ++j)
#pragma unroll
            for (int e = 0; e < 8; ++e) acc[mi][j][e] = 0.0f;

    for (int k0 = 0; k0 < K; k0 += 32) {
        if (AF32) tile_load_f32 (As, (const float *)Ap, m0, k0, K, tid);
        else      tile_load_bf16(As, (const __bf16*)Ap, m0, k0, K, tid);
        if (BF32) tile_load_f32 (Bs, (const float *)Bp, n0, k0, K, tid);
        else      tile_load_bf16(Bs, (const __bf16*)Bp, n0, k0, K, tid);

        // prefetch next K-tile into L2 (global_prefetch_b8)
        if (k0 + 32 < K) {
            const size_t eA = AF32 ? 4 : 2, eB = BF32 ? 4 : 2;
            __builtin_prefetch((const char*)Ap +
                ((m0 + (tid >> 1)) * (size_t)K + k0 + 32 + (tid & 1) * 16) * eA, 0, 1);
            __builtin_prefetch((const char*)Bp +
                ((n0 + (tid >> 1)) * (size_t)K + k0 + 32 + (tid & 1) * 16) * eB, 0, 1);
        }
        __syncthreads();

        v16bf a[2], b[4];
#pragma unroll
        for (int mi = 0; mi < 2; ++mi)
            a[mi] = ldfrag(&As[(wm + mi * 16 + r) * TSTR], kb);
#pragma unroll
        for (int j = 0; j < 4; ++j)
            b[j] = ldfrag(&Bs[(wn + j * 16 + r) * TSTR], kb);
#pragma unroll
        for (int mi = 0; mi < 2; ++mi)
#pragma unroll
            for (int j = 0; j < 4; ++j)
                acc[mi][j] = wmma_bf16(a[mi], b[j], acc[mi][j]);
        __syncthreads();
    }

    // C layout: VGPR i -> row (half*8 + i), col = lane&15 within 16x16 frag
    const int half = lane >> 4;
    const int cc   = lane & 15;
#pragma unroll
    for (int mi = 0; mi < 2; ++mi)
#pragma unroll
        for (int j = 0; j < 4; ++j)
#pragma unroll
            for (int i = 0; i < 8; ++i) {
                size_t row = m0 + wm + mi * 16 + half * 8 + i;
                size_t col = n0 + wn + j * 16 + cc;
                if (CF32) ((float *)Cp)[row * (size_t)N + col] = acc[mi][j][i];
                else      ((__bf16*)Cp)[row * (size_t)N + col] = (__bf16)acc[mi][j][i];
            }
}

// ---------------------------------------------------------------------------
// RoPE + QKV split.  qkv (bf16, S x 3E) -> Q,K (H,S,D) bf16 and Vt (H,D,S).
// ---------------------------------------------------------------------------
__global__ void __launch_bounds__(256)
rope_split_kernel(const __bf16* __restrict__ qkv, const float* __restrict__ cosb,
                  const float* __restrict__ sinb, __bf16* __restrict__ Q,
                  __bf16* __restrict__ Kb, __bf16* __restrict__ Vt) {
    const int s   = blockIdx.x;
    const int tid = threadIdx.x;
    const __bf16* row = qkv + (size_t)s * (3 * EMB);
#pragma unroll
    for (int t = 0; t < 8; ++t) {
        int p = tid + t * 256;       // 0..2047 : (head, d<64) pairs
        int h = p >> 6;
        int d = p & 63;
        float c  = cosb[s * HD + d];
        float sn = sinb[s * HD + d];
        int base = h * HD + d;
        size_t qo = (size_t)h * SEQ * HD + (size_t)s * HD + d;

        float q1 = (float)row[base], q2 = (float)row[base + 64];
        Q[qo]      = (__bf16)(q1 * c - q2 * sn);
        Q[qo + 64] = (__bf16)(q2 * c + q1 * sn);

        float k1 = (float)row[EMB + base], k2 = (float)row[EMB + base + 64];
        Kb[qo]      = (__bf16)(k1 * c - k2 * sn);
        Kb[qo + 64] = (__bf16)(k2 * c + k1 * sn);

        size_t vo = (size_t)h * HD * SEQ + (size_t)d * SEQ + s;
        Vt[vo]            = row[2 * EMB + base];
        Vt[vo + 64 * SEQ] = row[2 * EMB + base + 64];
    }
}

// ---------------------------------------------------------------------------
// Flash attention: one workgroup per (q-tile of 128, head). 8 waves,
// each wave owns 16 query rows. Tiles moved via async-to-LDS DMA.
// ---------------------------------------------------------------------------
constexpr int LDSP = 136;  // padded LDS row stride (elements, 16B-aligned)

__device__ __forceinline__ void async_tile_load(const __bf16* g_base, long long ldg,
                                                __bf16* lds_base, int tid) {
    // 128 rows x 128 bf16 cols = 2048 16-byte chunks; 8 per thread.
#pragma unroll
    for (int t = 0; t < 8; ++t) {
        int cidx = tid + t * 256;
        int row  = cidx >> 4;
        int c8   = (cidx & 15) << 3;
        unsigned lds_off = (unsigned)(unsigned long long)(lds_base + row * LDSP + c8);
        unsigned long long gaddr =
            (unsigned long long)(g_base + (long long)row * ldg + c8);
        asm volatile("global_load_async_to_lds_b128 %0, %1, off"
                     :: "v"(lds_off), "v"(gaddr) : "memory");
    }
}

__global__ void __launch_bounds__(256)
attn_kernel(const __bf16* __restrict__ Q, const __bf16* __restrict__ Kg,
            const __bf16* __restrict__ Vt, __bf16* __restrict__ attn) {
    extern __shared__ __bf16 sm[];
    __bf16* Qs = sm;                 // 128 x LDSP  (q rows, d cols)
    __bf16* Ks = Qs + 128 * LDSP;    // 128 x LDSP  (k rows, d cols)
    __bf16* Vs = Ks + 128 * LDSP;    // 128 x LDSP  (d rows, k cols)
    __bf16* Ps = Vs + 128 * LDSP;    // 128 x LDSP  (q rows, k cols)

    const int tid  = threadIdx.x;
    const int lane = tid & 31;
    const int wid  = tid >> 5;
    const int r    = lane & 15;
    const int kb   = (lane >> 4) * 8;
    const int half = lane >> 4;
    const int cc   = lane & 15;
    const int qt   = blockIdx.x;
    const int h    = blockIdx.y;
    const int qbase = qt * 128;
    const int qrow0 = wid * 16;

    // Q tile: loaded once per block via async DMA
    async_tile_load(Q + (size_t)h * SEQ * HD + (size_t)qbase * HD, HD, Qs, tid);
    asm volatile("s_wait_asynccnt 0x0" ::: "memory");
    __syncthreads();

    v8f o_acc[8];
    float m_i[8], l_i[8];
#pragma unroll
    for (int j = 0; j < 8; ++j)
#pragma unroll
        for (int e = 0; e < 8; ++e) o_acc[j][e] = 0.0f;
#pragma unroll
    for (int i = 0; i < 8; ++i) { m_i[i] = -1e30f; l_i[i] = 0.0f; }

    for (int kt = 0; kt <= qt; ++kt) {
        __syncthreads();  // previous iteration done with Ks/Vs
        async_tile_load(Kg + (size_t)h * SEQ * HD + (size_t)kt * 128 * HD, HD, Ks, tid);
        async_tile_load(Vt + (size_t)h * HD * SEQ + (size_t)kt * 128,      SEQ, Vs, tid);
        asm volatile("s_wait_asynccnt 0x0" ::: "memory");
        __syncthreads();

        // S = Q * K^T  (16 q rows x 128 k cols per wave)
        v8f s_acc[8];
#pragma unroll
        for (int j = 0; j < 8; ++j)
#pragma unroll
            for (int e = 0; e < 8; ++e) s_acc[j][e] = 0.0f;
#pragma unroll
        for (int kk = 0; kk < 128; kk += 32) {
            v16bf a = ldfrag(&Qs[(qrow0 + r) * LDSP + kk], kb);
#pragma unroll
            for (int j = 0; j < 8; ++j) {
                v16bf b = ldfrag(&Ks[(j * 16 + r) * LDSP + kk], kb);
                s_acc[j] = wmma_bf16(a, b, s_acc[j]);
            }
        }

        // scale + causal mask (only the diagonal tile needs masking)
        const bool diag = (kt == qt);
#pragma unroll
        for (int j = 0; j < 8; ++j)
#pragma unroll
            for (int i = 0; i < 8; ++i) {
                float sv = s_acc[j][i] * SCALE;
                if (diag && (j * 16 + cc) > (qrow0 + half * 8 + i - qrow0 + qrow0) - qrow0 + (half * 8 + i) - (half * 8 + i)) {}
                if (diag) {
                    int qg = half * 8 + i + qrow0;  // row within 128-q tile
                    int kg = j * 16 + cc;           // col within 128-k tile
                    if (kg > qg) sv = -1e30f;
                }
                s_acc[j][i] = sv;
            }

        // online softmax per query row (rows spread over 16-lane halves)
#pragma unroll
        for (int i = 0; i < 8; ++i) {
            float rm = -3.0e38f;
#pragma unroll
            for (int j = 0; j < 8; ++j) rm = fmaxf(rm, s_acc[j][i]);
#pragma unroll
            for (int off = 8; off >= 1; off >>= 1)
                rm = fmaxf(rm, __shfl_xor(rm, off, 32));
            float mn = fmaxf(m_i[i], rm);
            float al = exp2f((m_i[i] - mn) * LOG2E);
            float rs = 0.0f;
#pragma unroll
            for (int j = 0; j < 8; ++j) {
                float p = exp2f((s_acc[j][i] - mn) * LOG2E);
                rs += p;
                Ps[(qrow0 + half * 8 + i) * LDSP + j * 16 + cc] = (__bf16)p;
            }
#pragma unroll
            for (int off = 8; off >= 1; off >>= 1)
                rs += __shfl_xor(rs, off, 32);
            l_i[i] = l_i[i] * al + rs;
            m_i[i] = mn;
#pragma unroll
            for (int j2 = 0; j2 < 8; ++j2) o_acc[j2][i] *= al;
        }

        // wave-private P rows: ensure DS stores land before fragment reload
        asm volatile("s_wait_dscnt 0x0" ::: "memory");

        // O += P * V   (A = P 16x32-k frags, B = Vt 32-k x 16-d frags)
#pragma unroll
        for (int kk = 0; kk < 128; kk += 32) {
            v16bf a = ldfrag(&Ps[(qrow0 + r) * LDSP + kk], kb);
#pragma unroll
            for (int j2 = 0; j2 < 8; ++j2) {
                v16bf b = ldfrag(&Vs[(j2 * 16 + r) * LDSP + kk], kb);
                o_acc[j2] = wmma_bf16(a, b, o_acc[j2]);
            }
        }
    }

    // normalize and write attn (S, H*D) bf16
#pragma unroll
    for (int i = 0; i < 8; ++i) {
        float rl = 1.0f / l_i[i];
        size_t row = (size_t)qbase + qrow0 + half * 8 + i;
#pragma unroll
        for (int j2 = 0; j2 < 8; ++j2)
            attn[row * EMB + h * HD + j2 * 16 + cc] = (__bf16)(o_acc[j2][i] * rl);
    }
}

// ---------------------------------------------------------------------------
extern "C" void kernel_launch(void* const* d_in, const int* in_sizes, int n_in,
                              void* d_out, int out_size, void* d_ws, size_t ws_size,
                              hipStream_t stream) {
    const float* hidden = (const float*)d_in[0];
    const float* cosb   = (const float*)d_in[1];
    const float* sinb   = (const float*)d_in[2];
    const float* w_pack = (const float*)d_in[3];
    const float* w_o    = (const float*)d_in[4];
    float* out = (float*)d_out;

    // workspace layout (bf16):
    //   qkv  : S x 3E        = 50331648 B
    //   Q    : H x S x D     = 16777216 B
    //   K    : H x S x D     = 16777216 B
    //   Vt   : H x D x S     = 16777216 B
    //   attn : S x E         = 16777216 B
    char* ws = (char*)d_ws;
    __bf16* qkv  = (__bf16*)ws;
    __bf16* Qb   = (__bf16*)(ws + (size_t)SEQ * 3 * EMB * 2);
    __bf16* Kb   = Qb + (size_t)NH * SEQ * HD;
    __bf16* Vt   = Kb + (size_t)NH * SEQ * HD;
    __bf16* attn = Vt + (size_t)NH * SEQ * HD;

    // 1) QKV = hidden * w_pack^T  (f32 in, bf16 out)
    gemm_bt_kernel<true, true, false>
        <<<dim3(3 * EMB / 128, SEQ / 128), 256, 0, stream>>>(
            hidden, w_pack, qkv, SEQ, 3 * EMB, EMB);

    // 2) RoPE + split into Q, K, V^T
    rope_split_kernel<<<SEQ, 256, 0, stream>>>(qkv, cosb, sinb, Qb, Kb, Vt);

    // 3) causal flash attention
    const int shbytes = 4 * 128 * LDSP * 2;  // 139264 B dynamic LDS
    hipFuncSetAttribute(reinterpret_cast<const void*>(attn_kernel),
                        hipFuncAttributeMaxDynamicSharedMemorySize, shbytes);
    attn_kernel<<<dim3(SEQ / 128, NH), 256, shbytes, stream>>>(Qb, Kb, Vt, attn);

    // 4) out = attn * w_o^T  (bf16 x f32 -> f32)
    gemm_bt_kernel<false, true, true>
        <<<dim3(EMB / 128, SEQ / 128), 256, 0, stream>>>(
            attn, w_o, out, SEQ, EMB, EMB);
}
